// TemporalAttentiveFusionNet_46901042872663
// MI455X (gfx1250) — compile-verified, matrix operand order checked by reading, not run on hardware
//
#include <hip/hip_runtime.h>

// ---------------------------------------------------------------------------
// CDNA5 (gfx1250) fused TemporalAttentiveFusionNet
//   B=32768 rows, SEQ=24, E=128, H=4, HD=32, 5 series -> 120 tokens/row
//   One workgroup (256 threads = 8 wave32) per batch row.
//   All GEMM-shaped math via v_wmma_f32_16x16x32_f16.
//   Weight panels staged to LDS once per WG (8x cut in L2 traffic).
//   QKV loop fully unrolled: fragment loads batched ahead of WMMAs so the
//   scheduler can use partial s_wait_dscnt, and Q/K/V routing folds statically.
// ---------------------------------------------------------------------------

typedef _Float16 half8   __attribute__((ext_vector_type(8)));
typedef _Float16 half16  __attribute__((ext_vector_type(16)));
typedef float    floatx8 __attribute__((ext_vector_type(8)));

#define NROWS    32768
#define NTOK     120          // 5 series * 24 tokens
#define ESTRIDE  136          // padded f16 stride for 128-wide LDS buffers
#define VSTRIDE  40           // padded f16 stride for 32-wide (j) tiles

// LDS layout (halfs / floats), carved out of dynamic shared memory
#define HQ      (128 * ESTRIDE)        // Q  f16
#define HK      (128 * ESTRIDE)        // K  f16
#define HVT     (20 * 32 * VSTRIDE)    // V^T per (s,h): [d][j]
#define HAO     (128 * ESTRIDE)        // emb (phase 0/1) then attn-out (phase 2/3)
#define HATT    (8 * 32 * VSTRIDE)     // per-wave softmax scratch
#define HWGT    (192 * ESTRIDE)        // staged weight panel (half of wqkv / all of wout)
#define NHALF   (HQ + HK + HVT + HAO + HATT + HWGT)
#define NFLOAT  (640 + 128 + 128 + 128 + 384 + 8 + 64 + 16 + 16)
#define SMEM_BYTES (NHALF * 2 + NFLOAT * 4 + 128 * 4)        // + idx ints

// --------------------------- fragment helpers ------------------------------
// 16-bit A/B fragment (16x32 / 32x16) for v_wmma_*_16x16x32_*:
//  lane l (l<16):  element m(or n)=l,   K = {0..7} and {16..23}
//  lane l (l>=16): element m(or n)=l-16, K = {8..15} and {24..31}
__device__ __forceinline__ half16 load_frag(const _Float16* base, int stride, int lane) {
  const int r = lane & 15;
  const _Float16* p = base + r * stride + ((lane & 16) ? 8 : 0);
  half8 lo = *(const half8*)(p);
  half8 hi = *(const half8*)(p + 16);
  half16 f;
#pragma unroll
  for (int i = 0; i < 8; ++i) { f[i] = lo[i]; f[i + 8] = hi[i]; }
  return f;
}

__device__ __forceinline__ floatx8 wmma16(half16 a, half16 b, floatx8 c) {
  return __builtin_amdgcn_wmma_f32_16x16x32_f16(false, a, false, b, (short)0, c,
                                                false, false);
}

// copy a [rows x 128] f16 panel (row-major, stride 128) into LDS with stride 136
__device__ __forceinline__ void stage_panel(_Float16* dst, const _Float16* src,
                                            int rows, int tid) {
  for (int u = tid; u < rows * 16; u += 256) {
    int r = u >> 4, c = (u & 15) << 3;
    *(half8*)(dst + r * ESTRIDE + c) = *(const half8*)(src + r * 128 + c);
  }
}

// --------------------------- weight prep -----------------------------------
__global__ void prep_weights(const float* __restrict__ inw,   // [384][128]
                             const float* __restrict__ outw,  // [128][128]
                             _Float16* __restrict__ wqkv,
                             _Float16* __restrict__ wout) {
  int i = blockIdx.x * blockDim.x + threadIdx.x;
  if (i < 384 * 128) wqkv[i] = (_Float16)inw[i];
  if (i < 128 * 128) wout[i] = (_Float16)outw[i];
}

// --------------------------- fused main kernel -----------------------------
extern "C" __global__ __launch_bounds__(256)
void fused_tafn(const float* __restrict__ x,
                const float* __restrict__ emb_table,
                const float* __restrict__ in_proj_b,
                const float* __restrict__ out_proj_b,
                const float* __restrict__ ln_g,
                const float* __restrict__ ln_b,
                const float* __restrict__ attn_fc_w,
                const float* __restrict__ attn_fc_b,
                const float* __restrict__ fc1_w, const float* __restrict__ fc1_b,
                const float* __restrict__ fc2_w, const float* __restrict__ fc2_b,
                const float* __restrict__ fc3_w, const float* __restrict__ fc3_b,
                const float* __restrict__ fc4_w, const float* __restrict__ fc4_b,
                const _Float16* __restrict__ wqkv,   // [384][128] f16
                const _Float16* __restrict__ wout,   // [128][128] f16
                float* __restrict__ out) {
  extern __shared__ __align__(16) unsigned char smem[];
  _Float16* sQ    = (_Float16*)smem;
  _Float16* sK    = sQ + HQ;
  _Float16* sVT   = sK + HK;
  _Float16* sAo   = sVT + HVT;          // aliases: emb during QKV, attn-out after
  _Float16* attnS = sAo + HAO;
  _Float16* sWgt  = attnS + HATT;       // staged weight panel
  float* sPooled  = (float*)(sWgt + HWGT);    // 5*128
  float* sLNG     = sPooled + 640;
  float* sLNB     = sLNG + 128;
  float* sOB      = sLNB + 128;
  float* sIPB     = sOB + 128;          // 384: in_proj bias
  float* sL1      = sIPB + 384;         // 8
  float* sH1      = sL1 + 8;            // 64
  float* sH2      = sH1 + 64;           // 16
  float* sH3      = sH2 + 16;           // 16
  int*   sIdx     = (int*)(sH3 + 16);   // 120

  const int row  = blockIdx.x;
  const int tid  = threadIdx.x;
  const int wid  = tid >> 5;
  const int lane = tid & 31;
  const int cl   = lane & 15;
  const int hoff = (lane >= 16) ? 8 : 0;

  // ---------------- phase 0: init + embedding gather (f32 -> f16 LDS) ------
  for (int i = tid; i < 640; i += 256) sPooled[i] = 0.0f;
  for (int i = tid; i < 128; i += 256) {
    sLNG[i] = ln_g[i]; sLNB[i] = ln_b[i]; sOB[i] = out_proj_b[i];
  }
  for (int i = tid; i < 384; i += 256) sIPB[i] = in_proj_b[i];
  if (tid < NTOK) {
    float v = x[(size_t)row * 122 + 2 + tid];   // series laid out (s,t) contiguous
    int id = (int)rintf(v * 1000.0f) + 5000;
    id = min(max(id, 0), 9999);
    sIdx[tid] = id;
  }
  __builtin_prefetch(wqkv + (tid << 6), 0, 3);   // pull first weight panel toward L2/WGP
  __syncthreads();
  for (int i = tid; i < 128 * 128; i += 256) {
    int r = i >> 7, c = i & 127;
    _Float16 h = (r < NTOK) ? (_Float16)emb_table[(size_t)sIdx[r] * 128 + c]
                            : (_Float16)0.0f;
    sAo[r * ESTRIDE + c] = h;                    // sAo holds emb for now
  }

  // ---------------- phase 1: QKV GEMM  [128x128]x[128x384] -----------------
  // Weights staged through LDS in two 192-row chunks (96 KB total, loaded once
  // per workgroup instead of once per wave). Fully unrolled so Q/K/V routing
  // resolves at compile time and ds-loads pipeline ahead of the WMMAs.
  {
    half16 afr[4];
    bool afr_loaded = false;

#pragma unroll
    for (int chunk = 0; chunk < 2; ++chunk) {
      __syncthreads();                       // prior users of sWgt / sAo done
      stage_panel(sWgt, wqkv + chunk * 192 * 128, 192, tid);
      __syncthreads();

      if (!afr_loaded) {                     // emb fragments (sAo) valid now
#pragma unroll
        for (int ks = 0; ks < 4; ++ks)
          afr[ks] = load_frag(sAo + (wid * 16) * ESTRIDE + ks * 32, ESTRIDE, lane);
        afr_loaded = true;
      }

#pragma unroll
      for (int nt2 = 0; nt2 < 12; ++nt2) {
        if (chunk == 0)                       // prefetch second chunk while computing
          __builtin_prefetch(wqkv + 192 * 128 + nt2 * 2048, 0, 3);
        const int nt = chunk * 12 + nt2;

        half16 bf[4];
#pragma unroll
        for (int ks = 0; ks < 4; ++ks)
          bf[ks] = load_frag(sWgt + nt2 * 16 * ESTRIDE + ks * 32, ESTRIDE, lane);

        floatx8 c = {};
#pragma unroll
        for (int ks = 0; ks < 4; ++ks) c = wmma16(afr[ks], bf[ks], c);

        const int f = nt * 16 + cl;
        const float bias = sIPB[f];
#pragma unroll
        for (int j = 0; j < 8; ++j) {
          int r = wid * 16 + j + hoff;
          float v = c[j] + bias;
          if (f < 128) {
            sQ[r * ESTRIDE + f] = (_Float16)v;
          } else if (f < 256) {
            sK[r * ESTRIDE + (f - 128)] = (_Float16)v;
          } else if (r < NTOK) {
            int s = r / 24, t = r % 24;
            int hd = f - 256, h = hd >> 5, d = hd & 31;
            sVT[((s * 4 + h) * 32 + d) * VSTRIDE + t] = (_Float16)v;  // transposed
          }
        }
      }
    }
  }
  __builtin_prefetch(wout + (tid << 6), 0, 3);   // pull out-proj panel early
  __syncthreads();

  // ---------------- phase 2: attention per (series, head) ------------------
  {
    const float scale = 0.17677669529663687f;  // 1/sqrt(32)
    for (int p = wid; p < 20; p += 8) {
      const int s = p >> 2, h = p & 3;
      const _Float16* qb = sQ + (s * 24) * ESTRIDE + h * 32;
      const _Float16* kb = sK + (s * 24) * ESTRIDE + h * 32;
      half16 aq[2], bk[2];
#pragma unroll
      for (int mt = 0; mt < 2; ++mt) aq[mt] = load_frag(qb + mt * 16 * ESTRIDE, ESTRIDE, lane);
#pragma unroll
      for (int nt = 0; nt < 2; ++nt) bk[nt] = load_frag(kb + nt * 16 * ESTRIDE, ESTRIDE, lane);

      floatx8 sc[2][2];
#pragma unroll
      for (int mt = 0; mt < 2; ++mt)
#pragma unroll
        for (int nt = 0; nt < 2; ++nt) {
          floatx8 z = {};
          sc[mt][nt] = wmma16(aq[mt], bk[nt], z);
        }

      _Float16* as = attnS + wid * 32 * VSTRIDE;
      const bool v1 = (cl < 8);   // column 16+cl valid iff < 24
#pragma unroll
      for (int mt = 0; mt < 2; ++mt) {
#pragma unroll
        for (int j = 0; j < 8; ++j) {
          float s0 = sc[mt][0][j] * scale;
          float s1 = v1 ? sc[mt][1][j] * scale : -3.0e38f;
          float m = fmaxf(s0, s1);
#pragma unroll
          for (int off = 1; off < 16; off <<= 1) m = fmaxf(m, __shfl_xor(m, off, 32));
          float e0 = __expf(s0 - m);
          float e1 = v1 ? __expf(s1 - m) : 0.0f;
          float sum = e0 + e1;
#pragma unroll
          for (int off = 1; off < 16; off <<= 1) sum += __shfl_xor(sum, off, 32);
          float inv = 1.0f / sum;
          int r = mt * 16 + j + hoff;
          as[r * VSTRIDE + cl]      = (_Float16)(e0 * inv);
          as[r * VSTRIDE + 16 + cl] = (_Float16)(e1 * inv);
        }
      }

      // attn (24x32, zero-padded cols) x V^T  -> attn-out tile
      half16 ba[2], bv[2];
#pragma unroll
      for (int mt = 0; mt < 2; ++mt) ba[mt] = load_frag(as + mt * 16 * VSTRIDE, VSTRIDE, lane);
      const _Float16* vtb = sVT + (s * 4 + h) * 32 * VSTRIDE;
#pragma unroll
      for (int nt = 0; nt < 2; ++nt) bv[nt] = load_frag(vtb + nt * 16 * VSTRIDE, VSTRIDE, lane);
#pragma unroll
      for (int mt = 0; mt < 2; ++mt)
#pragma unroll
        for (int nt = 0; nt < 2; ++nt) {
          floatx8 z = {};
          floatx8 o = wmma16(ba[mt], bv[nt], z);
#pragma unroll
          for (int j = 0; j < 8; ++j) {
            int rl = mt * 16 + j + hoff;
            if (rl < 24)
              sAo[(s * 24 + rl) * ESTRIDE + h * 32 + nt * 16 + cl] = (_Float16)o[j];
          }
        }
    }
  }
  __syncthreads();

  // -------- phase 3: out_proj GEMM + bias + LayerNorm + pooled -------------
  // Stage out-proj weight panel (32 KB) into LDS once per workgroup.
  stage_panel(sWgt, wout, 128, tid);
  __syncthreads();
  {
    half16 afr[4];
#pragma unroll
    for (int ks = 0; ks < 4; ++ks)
      afr[ks] = load_frag(sAo + (wid * 16) * ESTRIDE + ks * 32, ESTRIDE, lane);

    floatx8 acc[8] = {};
#pragma unroll
    for (int nt = 0; nt < 8; ++nt) {
      half16 bf[4];
#pragma unroll
      for (int ks = 0; ks < 4; ++ks)
        bf[ks] = load_frag(sWgt + nt * 16 * ESTRIDE + ks * 32, ESTRIDE, lane);
#pragma unroll
      for (int ks = 0; ks < 4; ++ks) acc[nt] = wmma16(afr[ks], bf[ks], acc[nt]);
    }

#pragma unroll
    for (int j = 0; j < 8; ++j) {
      float v[8];
      float sum = 0.0f;
#pragma unroll
      for (int nt = 0; nt < 8; ++nt) {
        v[nt] = acc[nt][j] + sOB[nt * 16 + cl];
        sum += v[nt];
      }
#pragma unroll
      for (int off = 1; off < 16; off <<= 1) sum += __shfl_xor(sum, off, 32);
      float mu = sum * (1.0f / 128.0f);
      float var = 0.0f;
#pragma unroll
      for (int nt = 0; nt < 8; ++nt) { float d = v[nt] - mu; var += d * d; }
#pragma unroll
      for (int off = 1; off < 16; off <<= 1) var += __shfl_xor(var, off, 32);
      float rstd = rsqrtf(var * (1.0f / 128.0f) + 1e-5f);

      int r = wid * 16 + j + hoff;
      if (r < NTOK) {
        int s = r / 24;
#pragma unroll
        for (int nt = 0; nt < 8; ++nt) {
          int col = nt * 16 + cl;
          float y = (v[nt] - mu) * rstd * sLNG[col] + sLNB[col];
          atomicAdd(&sPooled[s * 128 + col], y * (1.0f / 24.0f));   // mean over t
        }
      }
    }
  }
  __syncthreads();

  // -------- phase 4: summaries + tiny MLP head -----------------------------
  if (tid < 5) {
    float a = attn_fc_b[0];
    for (int c = 0; c < 128; ++c) a += sPooled[tid * 128 + c] * attn_fc_w[c];
    sL1[2 + tid] = a;
  }
  if (tid == 5) {
    sL1[0] = x[(size_t)row * 122 + 0];
    sL1[1] = x[(size_t)row * 122 + 1];
  }
  __syncthreads();
  if (tid < 64) {
    float a = fc1_b[tid];
#pragma unroll
    for (int i = 0; i < 7; ++i) a += sL1[i] * fc1_w[tid * 7 + i];
    sH1[tid] = fmaxf(a, 0.0f);
  }
  __syncthreads();
  if (tid < 16) {
    float a = fc2_b[tid];
    for (int i = 0; i < 64; ++i) a += sH1[i] * fc2_w[tid * 64 + i];
    sH2[tid] = fmaxf(a, 0.0f);
  }
  __syncthreads();
  if (tid < 16) {
    float a = fc3_b[tid];
#pragma unroll
    for (int i = 0; i < 16; ++i) a += sH2[i] * fc3_w[tid * 16 + i];
    sH3[tid] = fmaxf(a, 0.0f);
  }
  __syncthreads();
  if (tid == 0) {
    float a = fc4_b[0];
#pragma unroll
    for (int i = 0; i < 16; ++i) a += sH3[i] * fc4_w[i];
    out[row] = a;
  }
}

// --------------------------- launcher --------------------------------------
extern "C" void kernel_launch(void* const* d_in, const int* in_sizes, int n_in,
                              void* d_out, int out_size, void* d_ws, size_t ws_size,
                              hipStream_t stream) {
  const float* x    = (const float*)d_in[0];
  const float* emb  = (const float*)d_in[1];
  const float* inw  = (const float*)d_in[2];
  const float* inb  = (const float*)d_in[3];
  const float* outw = (const float*)d_in[4];
  const float* outb = (const float*)d_in[5];
  const float* lng  = (const float*)d_in[6];
  const float* lnb  = (const float*)d_in[7];
  const float* afw  = (const float*)d_in[8];
  const float* afb  = (const float*)d_in[9];
  const float* f1w  = (const float*)d_in[10];
  const float* f1b  = (const float*)d_in[11];
  const float* f2w  = (const float*)d_in[12];
  const float* f2b  = (const float*)d_in[13];
  const float* f3w  = (const float*)d_in[14];
  const float* f3b  = (const float*)d_in[15];
  const float* f4w  = (const float*)d_in[16];
  const float* f4b  = (const float*)d_in[17];

  _Float16* wqkv = (_Float16*)d_ws;          // 384*128 f16
  _Float16* wout = wqkv + 384 * 128;         // 128*128 f16

  prep_weights<<<(384 * 128 + 255) / 256, 256, 0, stream>>>(inw, outw, wqkv, wout);

  fused_tafn<<<NROWS, 256, SMEM_BYTES, stream>>>(
      x, emb, inb, outb, lng, lnb, afw, afb,
      f1w, f1b, f2w, f2b, f3w, f3b, f4w, f4b,
      wqkv, wout, (float*)d_out);
}
